// GRU_D_75411035783284
// MI455X (gfx1250) — matrix operand order repaired and checked by reading, not compiled, not run on hardware
//
#include <hip/hip_runtime.h>
#include <math.h>

typedef __attribute__((ext_vector_type(16))) _Float16 v16h;
typedef __attribute__((ext_vector_type(8)))  _Float16 v8h;
typedef __attribute__((ext_vector_type(8)))  float    v8f;

constexpr int B_ = 256, C_ = 64, T_ = 256, H_ = 512;
constexpr int HS = 520;   // h/rh LDS stride (halves): 1040B rows -> 16B aligned, conflict-free
constexpr int XS = 72;    // xt LDS stride  (halves): 144B rows  -> 16B aligned, conflict-free

// ---------------- WMMA helpers (layouts per CDNA5 ISA 7.12.2, wave32) ----------------

__device__ __forceinline__ v8f wmma_f16(v16h a, v16h b, v8f c) {
  return __builtin_amdgcn_wmma_f32_16x16x32_f16(false, a, false, b, (short)0, c, false, false);
}

// A fragment 16x32 f16 from row-major source: lane (m=lane&15, half=lane>>4) reads
// two contiguous 16B chunks of row m: halves [half*8, +8) and [16+half*8, +8).
__device__ __forceinline__ v16h load_a_frag(const _Float16* A, int lda) {
  const int lane = threadIdx.x & 31;
  const _Float16* p = A + (size_t)(lane & 15) * lda + ((lane >> 4) * 8);
  v8h lo = *(const v8h*)p;
  v8h hi = *(const v8h*)(p + 16);
  return __builtin_shufflevector(lo, hi, 0, 1, 2, 3, 4, 5, 6, 7,
                                 8, 9, 10, 11, 12, 13, 14, 15);
}

// B fragment from pre-packed weights: P points at this ntile's [ktile][lane][16] block.
__device__ __forceinline__ v16h load_b_packed(const _Float16* P, int kt) {
  const int lane = threadIdx.x & 31;
  return *(const v16h*)(P + (((size_t)kt * 32 + lane) << 4));
}

// ---------------- weight repack: f32 row-major [K][N] -> f16 fragment order ----------------
// dst[((nt*KT + kt)*32 + lane)*16 + e] = src[(kt*32 + (lane>>4)*8 + (e&7) + (e>>3)*16)*N + nt*16 + (lane&15)]
__global__ void pack_b_frag(const float* __restrict__ src, _Float16* __restrict__ dst,
                            int K, int N) {
  int i = blockIdx.x * blockDim.x + threadIdx.x;
  if (i >= K * N) return;
  const int KT = K >> 5;
  const int e = i & 15;
  const int lane = (i >> 4) & 31;
  const int kt = (i >> 9) % KT;
  const int nt = (i >> 9) / KT;
  const int n = nt * 16 + (lane & 15);
  const int k = kt * 32 + (lane >> 4) * 8 + (e & 7) + (e >> 3) * 16;
  dst[i] = (_Float16)src[(size_t)k * N + n];
}

// ---------------- scans / elementwise ----------------

__global__ void scan_kernel(const float* __restrict__ x,
                            float* __restrict__ xe, _Float16* __restrict__ ivh) {
  int idx = blockIdx.x * blockDim.x + threadIdx.x;
  if (idx >= B_ * C_) return;
  const float* px = x + (size_t)idx * T_ * 3;
  float* pe = xe + (size_t)idx * T_;
  _Float16* pi = ivh + (size_t)idx * T_;
  float carry = px[0];
  float ivc = 1.0f;
  float prev_mask = 0.f;
  for (int t = 0; t < T_; ++t) {
    float raw = px[t * 3 + 0];
    float m   = px[t * 3 + 2];
    float ne = m * raw + (1.f - m) * carry;
    carry = ne;
    pe[t] = ne;
    float iv;
    if (t == 0) iv = 0.f;
    else { iv = 1.f + ivc * (1.f - prev_mask); ivc = iv; }
    pi[t] = (_Float16)iv;
    prev_mask = m;
  }
}

__global__ void mean_kernel(const float* __restrict__ xe, float* __restrict__ me) {
  int idx = blockIdx.x * blockDim.x + threadIdx.x;
  if (idx >= B_ * T_) return;
  int b = idx / T_, t = idx % T_;
  float s = 0.f;
  for (int c = 0; c < C_; ++c) s += xe[((size_t)(b * C_ + c)) * T_ + t];
  me[idx] = s * (1.0f / C_);
}

// out[m][n] = exp(-relu(A @ Bpacked + bias[n])); one wave per 16x16 tile.
__global__ void gemm_exp_relu(const _Float16* __restrict__ A, int lda,
                              const _Float16* __restrict__ Bp,
                              const float* __restrict__ bias,
                              float* __restrict__ out, int M, int N, int K) {
  const int wave = threadIdx.x >> 5;
  const int tile = blockIdx.x * (blockDim.x >> 5) + wave;
  const int tn = N >> 4;
  if (tile >= (M >> 4) * tn) return;
  const int tm = tile / tn, tc = tile % tn;
  const int KT = K >> 5;
  const _Float16* Ab = A + (size_t)tm * 16 * lda;
  const _Float16* Pnt = Bp + (size_t)tc * KT * 512;
  v8f acc = {};
  for (int kt = 0; kt < KT; ++kt) {
    v16h a = load_a_frag(Ab + kt * 32, lda);
    acc = wmma_f16(a, load_b_packed(Pnt, kt), acc);
  }
  const int lane = threadIdx.x & 31;
  const int n = tc * 16 + (lane & 15);
  const int mbase = tm * 16 + (lane >> 4) * 8;
  const float bn = bias[n];
#pragma unroll
  for (int r = 0; r < 8; ++r) {
    float v = acc[r] + bn;
    out[(size_t)(mbase + r) * N + n] = __expf(-fmaxf(v, 0.f));
  }
}

__global__ void xnew_kernel(const float* __restrict__ x, const float* __restrict__ xe,
                            const float* __restrict__ d1, const float* __restrict__ me,
                            _Float16* __restrict__ xn) {
  int idx = blockIdx.x * blockDim.x + threadIdx.x;
  if (idx >= B_ * C_ * T_) return;
  int b = idx / (C_ * T_);
  int rem = idx % (C_ * T_);
  int c = rem / T_, t = rem % T_;
  float raw = x[(size_t)idx * 3 + 0];
  float m   = x[(size_t)idx * 3 + 2];
  float d   = d1[idx];
  float v = m * raw + (1.f - m) * (d * xe[idx] + (1.f - d) * me[b * T_ + t]);
  xn[((size_t)b * T_ + t) * C_ + c] = (_Float16)v;
}

// ---------------- fused GRU-D recurrence + output projection ----------------
// 16 workgroups x 16 batch rows; 8 waves; wave w owns H-columns [64w, 64w+64).
// h in LDS; per step: async-stage x_t into LDS (overlapped with decay), z/r GEMMs,
// r*h staged in LDS, hp GEMM + combine, fused h@Wo+bo straight to d_out.
__global__ void __launch_bounds__(256, 1)
gru_kernel(const _Float16* __restrict__ xn,   // [B,T,C] f16 row-major
           const float* __restrict__ d2,      // [B,T]
           const _Float16* __restrict__ Wzp, const _Float16* __restrict__ Wrp,
           const _Float16* __restrict__ Whp,  // packed, K=64,  N=512
           const _Float16* __restrict__ Uzp, const _Float16* __restrict__ Urp,
           const _Float16* __restrict__ Uhp,  // packed, K=512, N=512
           const float* __restrict__ bz, const float* __restrict__ br,
           const float* __restrict__ bh,
           const _Float16* __restrict__ Wop,  // packed, K=512, N=64
           const float* __restrict__ bo,
           float* __restrict__ out) {         // [B,C,T]
  __shared__ __align__(16) _Float16 hbuf[16 * HS];
  __shared__ __align__(16) _Float16 rhbuf[16 * HS];
  __shared__ __align__(16) _Float16 xtbuf[16 * XS];
  const int tid = threadIdx.x;
  const int wave = tid >> 5;
  const int lane = tid & 31;
  const int b0 = blockIdx.x * 16;
  const int nlo = lane & 15;
  const int mhalf = (lane >> 4) * 8;

  for (int i = tid; i < 16 * HS; i += 256) {
    hbuf[i] = (_Float16)0.f;
    rhbuf[i] = (_Float16)0.f;
  }
  __syncthreads();

  const unsigned xt_lds_base = (unsigned)(uintptr_t)(void*)xtbuf;
  const unsigned long long xn_base = (unsigned long long)(uintptr_t)xn;

  for (int t = 0; t < T_; ++t) {
    // async-stage x_t tile [16 rows x 64 halves] into LDS (128 lanes x 16B each)
    if (tid < 128) {
      const int m = tid >> 3, ch = tid & 7;
      unsigned ldsoff = xt_lds_base + (unsigned)((m * XS + ch * 8) * 2);
      unsigned voff = (unsigned)(((unsigned)((b0 + m) * T_ + t) * C_ + ch * 8) * 2);
      asm volatile("global_load_async_to_lds_b128 %0, %1, %2"
                   :
                   : "v"(ldsoff), "v"(voff), "s"(xn_base)
                   : "memory");
    }

    // h = dec[b] * h (overlaps with the async x_t DMA)
    for (int i = tid; i < 16 * H_; i += 256) {
      int m = i >> 9, n = i & (H_ - 1);
      float dec = d2[(size_t)(b0 + m) * T_ + t];
      hbuf[m * HS + n] = (_Float16)((float)hbuf[m * HS + n] * dec);
    }
    asm volatile("s_wait_asynccnt 0x0" ::: "memory");
    __syncthreads();

    float zreg[4][8];
    // ---- z and r gates ----
    for (int i = 0; i < 4; ++i) {
      const int q = wave * 4 + i;
      const _Float16* wzB = Wzp + (size_t)q * (2 * 512);
      const _Float16* wrB = Wrp + (size_t)q * (2 * 512);
      const _Float16* uzB = Uzp + (size_t)q * (16 * 512);
      const _Float16* urB = Urp + (size_t)q * (16 * 512);
      v8f zacc = {}; v8f racc = {};
#pragma unroll
      for (int kt = 0; kt < 2; ++kt) {
        v16h a = load_a_frag(xtbuf + kt * 32, XS);
        zacc = wmma_f16(a, load_b_packed(wzB, kt), zacc);
        racc = wmma_f16(a, load_b_packed(wrB, kt), racc);
      }
      for (int kt = 0; kt < 16; ++kt) {
        v16h a = load_a_frag(hbuf + kt * 32, HS);
        zacc = wmma_f16(a, load_b_packed(uzB, kt), zacc);
        racc = wmma_f16(a, load_b_packed(urB, kt), racc);
      }
      const int n = q * 16 + nlo;
      const float bzn = bz[n], brn = br[n];
#pragma unroll
      for (int r = 0; r < 8; ++r) {
        const int m = mhalf + r;
        float z  = 1.f / (1.f + __expf(-(zacc[r] + bzn)));
        float rr = 1.f / (1.f + __expf(-(racc[r] + brn)));
        zreg[i][r] = z;
        rhbuf[m * HS + n] = (_Float16)(rr * (float)hbuf[m * HS + n]);
      }
    }
    __syncthreads();  // rh complete for all waves

    // ---- candidate hp + combine (each wave touches only its own h columns) ----
    for (int i = 0; i < 4; ++i) {
      const int q = wave * 4 + i;
      const _Float16* whB = Whp + (size_t)q * (2 * 512);
      const _Float16* uhB = Uhp + (size_t)q * (16 * 512);
      v8f hacc = {};
#pragma unroll
      for (int kt = 0; kt < 2; ++kt) {
        v16h a = load_a_frag(xtbuf + kt * 32, XS);
        hacc = wmma_f16(a, load_b_packed(whB, kt), hacc);
      }
      for (int kt = 0; kt < 16; ++kt) {
        v16h a = load_a_frag(rhbuf + kt * 32, HS);
        hacc = wmma_f16(a, load_b_packed(uhB, kt), hacc);
      }
      const int n = q * 16 + nlo;
      const float bhn = bh[n];
#pragma unroll
      for (int r = 0; r < 8; ++r) {
        const int m = mhalf + r;
        float hp = tanhf(hacc[r] + bhn);
        float ho = (float)hbuf[m * HS + n];
        float z = zreg[i][r];
        hbuf[m * HS + n] = (_Float16)((1.f - z) * ho + z * hp);
      }
    }
    __syncthreads();  // full h_new visible

    // ---- fused output: out[b, :, t] = h_new @ Wo + bo ----
    if (wave < 4) {
      const _Float16* woB = Wop + (size_t)wave * (16 * 512);
      v8f oacc = {};
      for (int kt = 0; kt < 16; ++kt) {
        v16h a = load_a_frag(hbuf + kt * 32, HS);
        oacc = wmma_f16(a, load_b_packed(woB, kt), oacc);
      }
      const int c = wave * 16 + nlo;
      const float boc = bo[c];
#pragma unroll
      for (int r = 0; r < 8; ++r) {
        const int m = mhalf + r;
        out[((size_t)(b0 + m) * C_ + c) * T_ + t] = oacc[r] + boc;
      }
    }
    __syncthreads();  // protect h before next step's decay
  }
}

// ---------------- launcher ----------------

extern "C" void kernel_launch(void* const* d_in, const int* in_sizes, int n_in,
                              void* d_out, int out_size, void* d_ws, size_t ws_size,
                              hipStream_t stream) {
  const float* x  = (const float*)d_in[0];
  const float* W1 = (const float*)d_in[1];
  const float* b1 = (const float*)d_in[2];
  const float* W2 = (const float*)d_in[3];
  const float* b2 = (const float*)d_in[4];
  const float* Wz = (const float*)d_in[5];
  const float* Wr = (const float*)d_in[6];
  const float* Wh = (const float*)d_in[7];
  const float* Uz = (const float*)d_in[8];
  const float* Ur = (const float*)d_in[9];
  const float* Uh = (const float*)d_in[10];
  const float* bz = (const float*)d_in[11];
  const float* br = (const float*)d_in[12];
  const float* bh = (const float*)d_in[13];
  const float* Wo = (const float*)d_in[14];
  const float* bo = (const float*)d_in[15];
  float* out = (float*)d_out;

  const size_t BCT = (size_t)B_ * C_ * T_;
  char* w = (char*)d_ws;
  float*     xe   = (float*)w;      w += BCT * 4;
  _Float16*  ivh  = (_Float16*)w;   w += BCT * 2;
  float*     me   = (float*)w;      w += (size_t)B_ * T_ * 4;
  float*     d1   = (float*)w;      w += BCT * 4;
  float*     d2m  = (float*)w;      w += (size_t)B_ * T_ * 4;
  _Float16*  xnb  = (_Float16*)w;   w += BCT * 2;
  _Float16*  W1p  = (_Float16*)w;   w += (size_t)T_ * T_ * 2;
  _Float16*  W2p  = (_Float16*)w;   w += (size_t)C_ * T_ * T_ * 2;
  _Float16*  Wzp  = (_Float16*)w;   w += (size_t)C_ * H_ * 2;
  _Float16*  Wrp  = (_Float16*)w;   w += (size_t)C_ * H_ * 2;
  _Float16*  Whp  = (_Float16*)w;   w += (size_t)C_ * H_ * 2;
  _Float16*  Uzp  = (_Float16*)w;   w += (size_t)H_ * H_ * 2;
  _Float16*  Urp  = (_Float16*)w;   w += (size_t)H_ * H_ * 2;
  _Float16*  Uhp  = (_Float16*)w;   w += (size_t)H_ * H_ * 2;
  _Float16*  Wop  = (_Float16*)w;   w += (size_t)H_ * C_ * 2;

  auto pack = [&](const float* s, _Float16* d, int K, int N) {
    pack_b_frag<<<(K * N + 255) / 256, 256, 0, stream>>>(s, d, K, N);
  };
  pack(W1, W1p, T_, T_);            // 256 x 256
  pack(W2, W2p, C_ * T_, T_);       // 16384 x 256
  pack(Wz, Wzp, C_, H_);            // 64 x 512
  pack(Wr, Wrp, C_, H_);
  pack(Wh, Whp, C_, H_);
  pack(Uz, Uzp, H_, H_);            // 512 x 512
  pack(Ur, Urp, H_, H_);
  pack(Uh, Uhp, H_, H_);
  pack(Wo, Wop, H_, C_);            // 512 x 64

  scan_kernel<<<(B_ * C_) / 256, 256, 0, stream>>>(x, xe, ivh);
  mean_kernel<<<(B_ * T_) / 256, 256, 0, stream>>>(xe, me);

  // d1: [B*C, T] = interval[B*C, T] @ W1[T, T]
  gemm_exp_relu<<<(B_ * C_ / 16) * (T_ / 16) / 8, 256, 0, stream>>>(
      ivh, T_, W1p, b1, d1, B_ * C_, T_, T_);
  // d2: [B, T] = interval[B, C*T] @ W2[C*T, T]
  gemm_exp_relu<<<(B_ / 16) * (T_ / 16) / 8, 256, 0, stream>>>(
      ivh, C_ * T_, W2p, b2, d2m, B_, T_, C_ * T_);

  xnew_kernel<<<(int)(BCT / 256), 256, 0, stream>>>(x, xe, d1, me, xnb);

  gru_kernel<<<B_ / 16, 256, 0, stream>>>(xnb, d2m, Wzp, Wrp, Whp, Uzp, Urp, Uhp,
                                          bz, br, bh, Wop, bo, out);
}
